// NPUAttentionKernel_30081950941776
// MI455X (gfx1250) — matrix operand order repaired and checked by reading, not compiled, not run on hardware
//
#include <hip/hip_runtime.h>

#define HIDDEN   2048
#define NHEADS   16
#define HDIM     128
#define BATCH    2
#define SEQ      2048
#define BQ       128      // q rows per workgroup (8 waves x 16)
#define TK       64       // keys per tile
#define NWAVES   8
#define KT_PAD   72       // Kt row length in bf16 (64 keys + 8 pad, keeps 16B row stride)
#define V_PAD    136      // V row length in bf16 (128 dims + 8 pad)
#define P_PAD    72       // P row length in bf16 (64 keys + 8 pad)
#define SCALE    0.08838834764831845f   // 1/sqrt(128)

typedef __attribute__((ext_vector_type(16))) __bf16 v16bf;
typedef __attribute__((ext_vector_type(8)))  float  v8f;

union BF16x16 {
    v16bf  v;
    __bf16 e[16];
    uint4  q[2];
};

// wave32-safe xor-shuffle (stays within 16-lane halves for mask < 16)
__device__ __forceinline__ float xor_bcast_f(float x, int mask, int lane) {
    int r = __builtin_amdgcn_ds_bpermute(((lane ^ mask) << 2), __float_as_int(x));
    return __int_as_float(r);
}

__global__ __launch_bounds__(256)
void flash_attn_bf16_wmma(const float* __restrict__ Q,
                          const float* __restrict__ K,
                          const float* __restrict__ V,
                          float* __restrict__ O) {
    __shared__ alignas(16) __bf16 sKt[HDIM][KT_PAD];     // K transposed: [d][key]
    __shared__ alignas(16) __bf16 sV[TK][V_PAD];          // V row-major: [key][d]
    __shared__ alignas(16) __bf16 sP[NWAVES][16][P_PAD];  // per-wave P staging

    const int tid  = threadIdx.x;
    const int lane = tid & 31;
    const int wave = tid >> 5;
    const int half = lane >> 4;   // 0: lanes 0-15, 1: lanes 16-31
    const int l16  = lane & 15;

    const int h = blockIdx.y;
    const int b = blockIdx.z;
    const int q0 = blockIdx.x * BQ + wave * 16;   // first q row of this wave

    const size_t head_off = (size_t)b * SEQ * HIDDEN + (size_t)h * HDIM;
    const float* __restrict__ qbase = Q + head_off;
    const float* __restrict__ kbase = K + head_off;
    const float* __restrict__ vbase = V + head_off;
    float*       __restrict__ obase = O + head_off;

    // ---- Q A-fragments (4 frags cover d = 0..127), per ISA 16-bit A layout ----
    v16bf qa[4];
    {
        const float* qrow = qbase + (size_t)(q0 + l16) * HIDDEN;
        #pragma unroll
        for (int f = 0; f < 4; ++f) {
            BF16x16 t;
            const int d0 = f * 32 + half * 8;
            #pragma unroll
            for (int j = 0; j < 8; ++j) t.e[j]     = (__bf16)qrow[d0 + j];
            #pragma unroll
            for (int j = 0; j < 8; ++j) t.e[8 + j] = (__bf16)qrow[d0 + 16 + j];
            qa[f] = t.v;
        }
    }

    // ---- accumulators ----
    v8f oacc[8];
    #pragma unroll
    for (int nd = 0; nd < 8; ++nd) oacc[nd] = (v8f){0.f,0.f,0.f,0.f,0.f,0.f,0.f,0.f};
    float mrun[8], lrun[8];
    #pragma unroll
    for (int r = 0; r < 8; ++r) { mrun[r] = -INFINITY; lrun[r] = 0.f; }

    const int NT = SEQ / TK;
    for (int kt = 0; kt < NT; ++kt) {
        const int key0 = kt * TK;

        __syncthreads();   // previous iteration done reading sKt/sV

        // ---- cooperative K (transposed) + V tile load, fp32 -> bf16 ----
        // 64*128 floats = 2048 float4 chunks / 256 threads = 8 chunks each
        #pragma unroll
        for (int i = 0; i < 8; ++i) {
            const int c   = tid + i * 256;
            const int key = c >> 5;            // (c*4)/128
            const int d   = (c & 31) << 2;
            const float4 fk = *(const float4*)(kbase + (size_t)(key0 + key) * HIDDEN + d);
            sKt[d + 0][key] = (__bf16)fk.x;
            sKt[d + 1][key] = (__bf16)fk.y;
            sKt[d + 2][key] = (__bf16)fk.z;
            sKt[d + 3][key] = (__bf16)fk.w;
            const float4 fv = *(const float4*)(vbase + (size_t)(key0 + key) * HIDDEN + d);
            sV[key][d + 0] = (__bf16)fv.x;
            sV[key][d + 1] = (__bf16)fv.y;
            sV[key][d + 2] = (__bf16)fv.z;
            sV[key][d + 3] = (__bf16)fv.w;
            if (kt + 1 < NT) {   // prefetch next tile
                __builtin_prefetch(kbase + (size_t)(key0 + TK + key) * HIDDEN + d, 0, 0);
                __builtin_prefetch(vbase + (size_t)(key0 + TK + key) * HIDDEN + d, 0, 0);
            }
        }
        __syncthreads();

        // ---- scores: S(16x64) = Q(16x128) * K^T, 4 n-blocks x 4 d-blocks ----
        v8f sc[4];
        #pragma unroll
        for (int n = 0; n < 4; ++n) sc[n] = (v8f){0.f,0.f,0.f,0.f,0.f,0.f,0.f,0.f};
        #pragma unroll
        for (int dblk = 0; dblk < 4; ++dblk) {
            #pragma unroll
            for (int n = 0; n < 4; ++n) {
                BF16x16 bk;   // B lane = key column (n*16+l16), K = {half*8..+7, 16+half*8..+7} of d-block
                #pragma unroll
                for (int s = 0; s < 16; ++s)
                    bk.e[s] = sKt[dblk * 32 + (s < 8 ? half * 8 + s : 16 + half * 8 + (s - 8))][n * 16 + l16];
                sc[n] = __builtin_amdgcn_wmma_f32_16x16x32_bf16(
                            false, qa[dblk], false, bk.v, (short)0, sc[n], false, false);
            }
        }

        // ---- online softmax (rows: M = r + 8*half, cols across 16-lane half) ----
        float corr[8];
        #pragma unroll
        for (int r = 0; r < 8; ++r) {
            float mv = -INFINITY;
            #pragma unroll
            for (int n = 0; n < 4; ++n) mv = fmaxf(mv, sc[n][r] * SCALE);
            #pragma unroll
            for (int msk = 1; msk < 16; msk <<= 1) mv = fmaxf(mv, xor_bcast_f(mv, msk, lane));
            const float mnew = fmaxf(mrun[r], mv);
            const float c    = __expf(mrun[r] - mnew);
            corr[r] = c;
            float rs = 0.f;
            #pragma unroll
            for (int n = 0; n < 4; ++n) {
                const float p = __expf(sc[n][r] * SCALE - mnew);
                sc[n][r] = p;
                rs += p;
            }
            #pragma unroll
            for (int msk = 1; msk < 16; msk <<= 1) rs += xor_bcast_f(rs, msk, lane);
            lrun[r] = lrun[r] * c + rs;
            mrun[r] = mnew;
        }
        #pragma unroll
        for (int nd = 0; nd < 8; ++nd)
            #pragma unroll
            for (int r = 0; r < 8; ++r) oacc[nd][r] *= corr[r];

        // ---- stage P as bf16 into LDS (D-layout -> memory row-major) ----
        #pragma unroll
        for (int n = 0; n < 4; ++n)
            #pragma unroll
            for (int r = 0; r < 8; ++r)
                sP[wave][r + 8 * half][n * 16 + l16] = (__bf16)sc[n][r];

        __syncthreads();

        // ---- O(16x128) += P(16x64) * V(64x128): 2 k-chunks x 8 d-blocks ----
        #pragma unroll
        for (int kc = 0; kc < 2; ++kc) {
            BF16x16 pa;   // A layout: row = l16, K = {half*8..+7, 16+half*8..+7} of chunk
            const __bf16* prow = &sP[wave][l16][kc * 32 + half * 8];
            pa.q[0] = *(const uint4*)(prow);
            pa.q[1] = *(const uint4*)(prow + 16);
            #pragma unroll
            for (int nd = 0; nd < 8; ++nd) {
                BF16x16 bv;   // B lane = dim column (nd*16+l16), K = {half*8..+7, 16+half*8..+7} of chunk
                #pragma unroll
                for (int s = 0; s < 16; ++s)
                    bv.e[s] = sV[kc * 32 + (s < 8 ? half * 8 + s : 16 + half * 8 + (s - 8))][nd * 16 + l16];
                oacc[nd] = __builtin_amdgcn_wmma_f32_16x16x32_bf16(
                               false, pa.v, false, bv.v, (short)0, oacc[nd], false, false);
            }
        }
    }

    // ---- epilogue: normalize and store fp32 ----
    #pragma unroll
    for (int r = 0; r < 8; ++r) {
        const float inv = 1.0f / lrun[r];
        const int row = q0 + r + 8 * half;
        float* orow = obase + (size_t)row * HIDDEN;
        #pragma unroll
        for (int nd = 0; nd < 8; ++nd)
            orow[nd * 16 + l16] = oacc[nd][r] * inv;
    }
}

extern "C" void kernel_launch(void* const* d_in, const int* in_sizes, int n_in,
                              void* d_out, int out_size, void* d_ws, size_t ws_size,
                              hipStream_t stream) {
    const float* q = (const float*)d_in[0];
    const float* k = (const float*)d_in[1];
    const float* v = (const float*)d_in[2];
    float* out = (float*)d_out;
    (void)in_sizes; (void)n_in; (void)out_size; (void)d_ws; (void)ws_size;

    dim3 grid(SEQ / BQ, NHEADS, BATCH);   // (16, 16, 2)
    dim3 block(256);                       // 8 wave32 waves
    flash_attn_bf16_wmma<<<grid, block, 0, stream>>>(q, k, v, out);
}
